// TreeLSTM_39247411151311
// MI455X (gfx1250) — matrix operand order repaired and checked by reading, not compiled, not run on hardware
//
#include <hip/hip_runtime.h>
#include <hip/hip_bf16.h>

// ---------- types ----------
typedef __bf16 bf16_t;
typedef __attribute__((ext_vector_type(16))) __bf16 v16bf;
typedef __attribute__((ext_vector_type(8)))  __bf16 v8bf;
typedef __attribute__((ext_vector_type(8)))  float   v8f;
typedef __attribute__((ext_vector_type(4)))  float   v4f;

#define D_LVL   14
#define NNODES  16383      // 2^14 - 1
#define NLEAF   8192
#define LEAF0   8191       // first leaf index
#define HID     128
#define NIN     1024

// ---------- helpers ----------
__device__ __forceinline__ bf16_t f2bf(float f) {
    return (bf16_t)f;                       // native v_cvt on gfx1250
}
__device__ __forceinline__ unsigned short f2bf_u16(float f) {
    return __builtin_bit_cast(unsigned short, (bf16_t)f);
}
__device__ __forceinline__ float sigmoidf_(float x) {
    return 1.0f / (1.0f + __expf(-x));
}

// A fragment (16x32 bf16, row-major source, ld in elements).
// lanes 0-15: row=lane, K {0..7,16..23}; lanes 16-31: row=lane-16, K {8..15,24..31}
__device__ __forceinline__ v16bf load_a_bf16(const bf16_t* base, int ld, int lane) {
    int row  = lane & 15;
    int koff = (lane & 16) ? 8 : 0;
    const bf16_t* p = base + row * ld + koff;
    v8bf lo = *reinterpret_cast<const v8bf*>(p);
    v8bf hi = *reinterpret_cast<const v8bf*>(p + 16);
    v16bf r;
#pragma unroll
    for (int i = 0; i < 8; ++i) { r[i] = lo[i]; r[i + 8] = hi[i]; }
    return r;
}

// A fragment from f32 source with on-the-fly bf16 conversion
__device__ __forceinline__ v16bf load_a_f32(const float* base, int ld, int lane) {
    int row  = lane & 15;
    int koff = (lane & 16) ? 8 : 0;
    const float* p = base + row * ld + koff;
    v4f a0 = *reinterpret_cast<const v4f*>(p);
    v4f a1 = *reinterpret_cast<const v4f*>(p + 4);
    v4f b0 = *reinterpret_cast<const v4f*>(p + 16);
    v4f b1 = *reinterpret_cast<const v4f*>(p + 20);
    v16bf r;
#pragma unroll
    for (int i = 0; i < 4; ++i) {
        r[i]      = f2bf(a0[i]);
        r[i + 4]  = f2bf(a1[i]);
        r[i + 8]  = f2bf(b0[i]);
        r[i + 12] = f2bf(b1[i]);
    }
    return r;
}

// B fragment (32x16 bf16) from row-major W[out][in] (ld in elements).
// lane n (0-15): col n, K {0..7,16..23}; lanes 16-31: col n-16, K {8..15,24..31}.
__device__ __forceinline__ v16bf load_b_bf16(const bf16_t* base, int ld, int lane) {
    int col  = lane & 15;
    int koff = (lane & 16) ? 8 : 0;
    const bf16_t* p = base + col * ld + koff;
    v8bf lo = *reinterpret_cast<const v8bf*>(p);
    v8bf hi = *reinterpret_cast<const v8bf*>(p + 16);
    v16bf r;
#pragma unroll
    for (int i = 0; i < 8; ++i) { r[i] = lo[i]; r[i + 8] = hi[i]; }
    return r;
}

__device__ __forceinline__ v8f wmma_bf16(v16bf a, v16bf b, v8f c) {
    return __builtin_amdgcn_wmma_f32_16x16x32_bf16(false, a, false, b,
                                                   (short)0, c, false, false);
}

// C fragment store (16x16 f32): VGPR r -> row r (lanes 0-15) / r+8 (lanes 16-31)
__device__ __forceinline__ void store_c_f32(float* base, int ld, int lane, v8f c) {
    int col = lane & 15;
    int r0  = (lane & 16) ? 8 : 0;
#pragma unroll
    for (int r = 0; r < 8; ++r) base[(r0 + r) * ld + col] = c[r];
}
__device__ __forceinline__ void store_c_bf16(unsigned short* base, int ld, int lane, v8f c) {
    int col = lane & 15;
    int r0  = (lane & 16) ? 8 : 0;
#pragma unroll
    for (int r = 0; r < 8; ++r) base[(r0 + r) * ld + col] = f2bf_u16(c[r]);
}

__device__ __forceinline__ v8f zero8() {
    v8f z;
#pragma unroll
    for (int i = 0; i < 8; ++i) z[i] = 0.0f;
    return z;
}

// ---------- kernels ----------

// vectorized f32 -> bf16 convert, 8 elements/thread (n must be multiple of 8)
__global__ void k_cvt_bf16(const float* __restrict__ src,
                           unsigned short* __restrict__ dst, int n) {
    int i = (blockIdx.x * blockDim.x + threadIdx.x) * 8;
    if (i < n) {
        v4f a = *reinterpret_cast<const v4f*>(src + i);
        v4f b = *reinterpret_cast<const v4f*>(src + i + 4);
        union { unsigned short u[8]; v4f v; } o;
#pragma unroll
        for (int j = 0; j < 4; ++j) { o.u[j] = f2bf_u16(a[j]); o.u[4 + j] = f2bf_u16(b[j]); }
        *reinterpret_cast<v4f*>(dst + i) = o.v;
    }
}

// embeds(leaves) = feat_leaf @ W_emb^T : [8192,1024] x [128,1024]^T -> bf16 [8192,128]
// grid 256 x 128 threads; 32 rows per block; wave = (mt, 4 consecutive nt tiles)
// Double-buffered K loop: next iteration's A/B loads issue before current WMMAs.
__global__ void k_embed(const bf16_t* __restrict__ featbf,
                        const bf16_t* __restrict__ Wembbf,
                        unsigned short* __restrict__ embbf) {
    int lane = threadIdx.x & 31, wave = threadIdx.x >> 5;
    int mt = wave & 1, g = wave >> 1;
    int rowbase = blockIdx.x * 32 + mt * 16;
    const bf16_t* Abase = featbf + (size_t)rowbase * NIN;
    const bf16_t* Bbase = Wembbf + (size_t)(g * 64) * NIN;   // 4 N-tiles of 16 rows
    v8f acc[4];
#pragma unroll
    for (int j = 0; j < 4; ++j) acc[j] = zero8();

    v16bf a_cur = load_a_bf16(Abase, NIN, lane);
    v16bf b_cur[4];
#pragma unroll
    for (int j = 0; j < 4; ++j)
        b_cur[j] = load_b_bf16(Bbase + (size_t)(j * 16) * NIN, NIN, lane);

#pragma unroll 2
    for (int ks = 0; ks < NIN / 32 - 1; ++ks) {
        v16bf a_nxt = load_a_bf16(Abase + (ks + 1) * 32, NIN, lane);
        v16bf b_nxt[4];
#pragma unroll
        for (int j = 0; j < 4; ++j)
            b_nxt[j] = load_b_bf16(Bbase + (size_t)(j * 16) * NIN + (ks + 1) * 32, NIN, lane);
#pragma unroll
        for (int j = 0; j < 4; ++j)
            acc[j] = wmma_bf16(a_cur, b_cur[j], acc[j]);
        a_cur = a_nxt;
#pragma unroll
        for (int j = 0; j < 4; ++j) b_cur[j] = b_nxt[j];
    }
#pragma unroll
    for (int j = 0; j < 4; ++j)
        acc[j] = wmma_bf16(a_cur, b_cur[j], acc[j]);

#pragma unroll
    for (int j = 0; j < 4; ++j)
        store_c_bf16(embbf + (size_t)rowbase * HID + (g * 4 + j) * 16, HID, lane, acc[j]);
}

// iou(leaves) = embeds @ W_iou^T, then gate/apply for leaves -> h, c
// grid 512 x 128; 16 leaf rows per block; wave owns 6 N-tiles
__global__ void k_leaf(const bf16_t* __restrict__ embbf,
                       const bf16_t* __restrict__ Wioubf,
                       const float* __restrict__ b_iou,
                       float* __restrict__ h, float* __restrict__ c) {
    __shared__ float iou[16][388];
    int lane = threadIdx.x & 31, wave = threadIdx.x >> 5;
    int rowbase = blockIdx.x * 16;
    const bf16_t* Abase = embbf + (size_t)rowbase * HID;
    const bf16_t* Bbase = Wioubf + (size_t)(wave * 6 * 16) * HID;
    v8f acc[6];
#pragma unroll
    for (int j = 0; j < 6; ++j) acc[j] = zero8();
#pragma unroll
    for (int ks = 0; ks < 4; ++ks) {
        v16bf a = load_a_bf16(Abase + ks * 32, HID, lane);
        v16bf bfr[6];
#pragma unroll
        for (int j = 0; j < 6; ++j)
            bfr[j] = load_b_bf16(Bbase + (size_t)(j * 16) * HID + ks * 32, HID, lane);
#pragma unroll
        for (int j = 0; j < 6; ++j)
            acc[j] = wmma_bf16(a, bfr[j], acc[j]);
    }
#pragma unroll
    for (int j = 0; j < 6; ++j)
        store_c_f32(&iou[0][0] + (wave * 6 + j) * 16, 388, lane, acc[j]);
    __syncthreads();
    for (int e = threadIdx.x; e < 16 * HID; e += 128) {
        int r = e >> 7, cc = e & 127;
        float iv = iou[r][cc]        + b_iou[cc];
        float ov = iou[r][128 + cc]  + b_iou[128 + cc];
        float uv = iou[r][256 + cc]  + b_iou[256 + cc];
        float cn = sigmoidf_(iv) * tanhf(uv);      // c_red = c0 = 0
        float hn = sigmoidf_(ov) * tanhf(cn);
        size_t g = (size_t)(LEAF0 + rowbase + r) * HID + cc;
        c[g] = cn;
        h[g] = hn;
    }
}

// one tree level: f = sigmoid(h_ch @ Ufw^T + bf); c_red = pairsum(f*c_ch);
// h_tild = pairsum(h_ch); iou = h_tild @ Uiou^T; gate/apply -> h,c at parents
// grid = ceil(MP/16) x 128 threads; 16 parents (32 children) per block
__global__ void k_level(float* __restrict__ h, float* __restrict__ c,
                        const bf16_t* __restrict__ Ufwbf,
                        const float* __restrict__ U_f_b,
                        const bf16_t* __restrict__ Uioubf,
                        const float* __restrict__ b_iou,
                        int P0, int MP) {
    __shared__ float  fc[32][132];
    __shared__ bf16_t htld[16][136];
    __shared__ float  iou[16][388];
    int lane = threadIdx.x & 31, wave = threadIdx.x >> 5;
    int pbase = blockIdx.x * 16;
    int crow  = 2 * P0 + 1 + 2 * pbase;       // first child row of this block

    // h_tild (pairwise child row sums) -> LDS bf16
    for (int e = threadIdx.x; e < 16 * HID; e += 128) {
        int r = e >> 7, cc = e & 127;
        float s = h[(size_t)(crow + 2 * r) * HID + cc] +
                  h[(size_t)(crow + 2 * r + 1) * HID + cc];
        htld[r][cc] = f2bf(s);
    }

    // forget-gate GEMM over 32 child rows, then fc = sigmoid(f)*c_ch -> LDS
    {
        int mt = wave & 1, g = wave >> 1;
        const float* Abase = h + (size_t)(crow + mt * 16) * HID;
        const bf16_t* Bbase = Ufwbf + (size_t)(g * 64) * HID;
        v8f acc[4];
#pragma unroll
        for (int j = 0; j < 4; ++j) acc[j] = zero8();
#pragma unroll
        for (int ks = 0; ks < 4; ++ks) {
            v16bf a = load_a_f32(Abase + ks * 32, HID, lane);
            v16bf bfr[4];
#pragma unroll
            for (int j = 0; j < 4; ++j)
                bfr[j] = load_b_bf16(Bbase + (size_t)(j * 16) * HID + ks * 32, HID, lane);
#pragma unroll
            for (int j = 0; j < 4; ++j)
                acc[j] = wmma_bf16(a, bfr[j], acc[j]);
        }
        int col = lane & 15;
        int r0  = (lane & 16) ? 8 : 0;
#pragma unroll
        for (int j = 0; j < 4; ++j) {
            int n = (g * 4 + j) * 16 + col;
            float bias = U_f_b[n];
#pragma unroll
            for (int r = 0; r < 8; ++r) {
                int childloc = mt * 16 + r0 + r;
                float fv = sigmoidf_(acc[j][r] + bias);
                float cv = c[(size_t)(crow + childloc) * HID + n];
                fc[childloc][n] = fv * cv;
            }
        }
    }
    __syncthreads();

    // iou = h_tild @ Uiou^T -> LDS ; wave owns 6 N-tiles
    {
        const bf16_t* Bbase = Uioubf + (size_t)(wave * 6 * 16) * HID;
        v8f acc[6];
#pragma unroll
        for (int j = 0; j < 6; ++j) acc[j] = zero8();
#pragma unroll
        for (int ks = 0; ks < 4; ++ks) {
            v16bf a = load_a_bf16(&htld[0][0] + ks * 32, 136, lane);
            v16bf bfr[6];
#pragma unroll
            for (int j = 0; j < 6; ++j)
                bfr[j] = load_b_bf16(Bbase + (size_t)(j * 16) * HID + ks * 32, HID, lane);
#pragma unroll
            for (int j = 0; j < 6; ++j)
                acc[j] = wmma_bf16(a, bfr[j], acc[j]);
        }
#pragma unroll
        for (int j = 0; j < 6; ++j)
            store_c_f32(&iou[0][0] + (wave * 6 + j) * 16, 388, lane, acc[j]);
    }
    __syncthreads();

    // gate + apply at parents
    for (int e = threadIdx.x; e < 16 * HID; e += 128) {
        int r = e >> 7, cc = e & 127;
        int p = pbase + r;
        if (p < MP) {
            float cred = fc[2 * r][cc] + fc[2 * r + 1][cc];
            float iv = iou[r][cc]       + b_iou[cc];
            float ov = iou[r][128 + cc] + b_iou[128 + cc];
            float uv = iou[r][256 + cc] + b_iou[256 + cc];
            float cn = sigmoidf_(iv) * tanhf(uv) + cred;
            float hn = sigmoidf_(ov) * tanhf(cn);
            size_t g = (size_t)(P0 + p) * HID + cc;
            c[g] = cn;
            h[g] = hn;
        }
    }
}

// logits = h @ W_lin^T + b_lin : [16383,128] -> d_out
// grid 512 x 128; 32 rows per block
__global__ void k_logits(const float* __restrict__ h,
                         const bf16_t* __restrict__ Wlinbf,
                         const float* __restrict__ b_lin,
                         float* __restrict__ out) {
    int lane = threadIdx.x & 31, wave = threadIdx.x >> 5;
    int mt = wave & 1, g = wave >> 1;
    int rowbase = blockIdx.x * 32 + mt * 16;
    const float*  Abase = h + (size_t)rowbase * HID;
    const bf16_t* Bbase = Wlinbf + (size_t)(g * 64) * HID;
    v8f acc[4];
#pragma unroll
    for (int j = 0; j < 4; ++j) acc[j] = zero8();
#pragma unroll
    for (int ks = 0; ks < 4; ++ks) {
        v16bf a = load_a_f32(Abase + ks * 32, HID, lane);
        v16bf bfr[4];
#pragma unroll
        for (int j = 0; j < 4; ++j)
            bfr[j] = load_b_bf16(Bbase + (size_t)(j * 16) * HID + ks * 32, HID, lane);
#pragma unroll
        for (int j = 0; j < 4; ++j)
            acc[j] = wmma_bf16(a, bfr[j], acc[j]);
    }
    int col = lane & 15;
    int r0  = (lane & 16) ? 8 : 0;
#pragma unroll
    for (int j = 0; j < 4; ++j) {
        int n = (g * 4 + j) * 16 + col;
        float bias = b_lin[n];
#pragma unroll
        for (int r = 0; r < 8; ++r) {
            int gg = rowbase + r0 + r;
            if (gg < NNODES) out[(size_t)gg * HID + n] = acc[j][r] + bias;
        }
    }
}

// ---------- launch ----------
extern "C" void kernel_launch(void* const* d_in, const int* in_sizes, int n_in,
                              void* d_out, int out_size, void* d_ws, size_t ws_size,
                              hipStream_t stream) {
    (void)in_sizes; (void)n_in; (void)out_size; (void)ws_size;
    const float* feat  = (const float*)d_in[0];
    const float* W_emb = (const float*)d_in[7];
    const float* W_iou = (const float*)d_in[8];
    const float* U_iou = (const float*)d_in[9];
    const float* b_iou = (const float*)d_in[10];
    const float* U_f_w = (const float*)d_in[11];
    const float* U_f_b = (const float*)d_in[12];
    const float* W_lin = (const float*)d_in[13];
    const float* b_lin = (const float*)d_in[14];
    float* out = (float*)d_out;

    char* ws = (char*)d_ws;
    const size_t OFF_FEATBF = 0;                                        // 8192*1024*2
    const size_t OFF_WEMB   = OFF_FEATBF + (size_t)NLEAF * NIN * 2;
    const size_t OFF_WIOU   = OFF_WEMB + (size_t)HID * NIN * 2;
    const size_t OFF_UIOU   = OFF_WIOU + (size_t)3 * HID * HID * 2;
    const size_t OFF_UFW    = OFF_UIOU + (size_t)3 * HID * HID * 2;
    const size_t OFF_WLIN   = OFF_UFW + (size_t)HID * HID * 2;
    const size_t OFF_EMB    = OFF_WLIN + (size_t)HID * HID * 2;
    const size_t OFF_H      = OFF_EMB + (size_t)NLEAF * HID * 2;
    const size_t OFF_C      = OFF_H + (size_t)16384 * HID * 4;

    unsigned short* featbf = (unsigned short*)(ws + OFF_FEATBF);
    unsigned short* Wembbf = (unsigned short*)(ws + OFF_WEMB);
    unsigned short* Wioubf = (unsigned short*)(ws + OFF_WIOU);
    unsigned short* Uioubf = (unsigned short*)(ws + OFF_UIOU);
    unsigned short* Ufwbf  = (unsigned short*)(ws + OFF_UFW);
    unsigned short* Wlinbf = (unsigned short*)(ws + OFF_WLIN);
    unsigned short* embbf  = (unsigned short*)(ws + OFF_EMB);
    float* h = (float*)(ws + OFF_H);
    float* c = (float*)(ws + OFF_C);

    auto cvt = [&](const float* s, unsigned short* d, int n) {
        int threads = n / 8;
        k_cvt_bf16<<<(threads + 255) / 256, 256, 0, stream>>>(s, d, n);
    };
    cvt(feat + (size_t)LEAF0 * NIN, featbf, NLEAF * NIN);   // leaf features only
    cvt(W_emb, Wembbf, HID * NIN);
    cvt(W_iou, Wioubf, 3 * HID * HID);
    cvt(U_iou, Uioubf, 3 * HID * HID);
    cvt(U_f_w, Ufwbf, HID * HID);
    cvt(W_lin, Wlinbf, HID * HID);

    k_embed<<<NLEAF / 32, 128, 0, stream>>>((const bf16_t*)featbf,
                                            (const bf16_t*)Wembbf, embbf);
    k_leaf<<<NLEAF / 16, 128, 0, stream>>>((const bf16_t*)embbf,
                                           (const bf16_t*)Wioubf, b_iou, h, c);

    for (int lvl = 1; lvl < D_LVL; ++lvl) {
        int P0 = (1 << (13 - lvl)) - 1;
        int MP = 1 << (13 - lvl);
        int blocks = (MP + 15) / 16;
        k_level<<<blocks, 128, 0, stream>>>(h, c, (const bf16_t*)Ufwbf, U_f_b,
                                            (const bf16_t*)Uioubf, b_iou, P0, MP);
    }

    k_logits<<<512, 128, 0, stream>>>(h, (const bf16_t*)Wlinbf, b_lin, out);
}
